// RotaryAttention_37984690766101
// MI455X (gfx1250) — matrix operand order repaired and checked by reading, not compiled, not run on hardware
//
#include <hip/hip_runtime.h>
#include <hip/hip_bf16.h>

typedef __attribute__((ext_vector_type(16))) _Float16 v16h;
typedef __attribute__((ext_vector_type(8)))  float    v8f;

#define HIDDEN 1024
#define HEADS 16
#define HEAD_DIM 64
#define BATCH 2
#define SEQ 2048
#define NROWS (BATCH * SEQ)                                  // 4096
#define PER_T ((size_t)BATCH * HEADS * SEQ * HEAD_DIM)       // 4194304 elems

#define WMMA(a, b, c) \
  __builtin_amdgcn_wmma_f32_16x16x32_f16(false, (a), false, (b), (short)0, (c), false, false)

#define ZERO8 {0.f, 0.f, 0.f, 0.f, 0.f, 0.f, 0.f, 0.f}

// ---------------- CDNA5 async global->LDS helpers ----------------
__device__ __forceinline__ void async_cp16(const _Float16* g, _Float16* l) {
#if defined(__gfx1250__) && __has_builtin(__builtin_amdgcn_global_load_async_to_lds_b128)
  typedef int vi4 __attribute__((vector_size(16)));
  typedef __attribute__((address_space(1))) vi4* gp_t;
  typedef __attribute__((address_space(3))) vi4* lp_t;
  __builtin_amdgcn_global_load_async_to_lds_b128(
      (gp_t)(unsigned long long)(uintptr_t)g, (lp_t)l, 0, 0);
#else
  *(float4*)l = *(const float4*)g;
#endif
}

__device__ __forceinline__ void async_wait0() {
#if defined(__gfx1250__) && __has_builtin(__builtin_amdgcn_s_wait_asynccnt)
  __builtin_amdgcn_s_wait_asynccnt(0);
#elif defined(__gfx1250__)
  asm volatile("s_wait_asynccnt 0x0" ::: "memory");
#endif
}

// ---- fragment loaders (contiguous k order k = 16*(lane/16) + s, wave32) ----

// A 16x32 f16 tile at p+kofs (row-major, 'stride' halves per row).
// lane<16: h[0..15]=K0..15 ; lane>=16: K16..31 ; M=lane%16
__device__ __forceinline__ v16h frag_a_lds(const _Float16* p, int stride, int kofs, int lane) {
  const int m  = lane & 15;
  const int ks = (lane >> 4) << 4;  // 0 or 16
  const _Float16* row = p + m * stride + kofs;
  v16h a;
#pragma unroll
  for (int i = 0; i < 8; ++i) a[i] = row[ks + i];
#pragma unroll
  for (int i = 0; i < 8; ++i) a[8 + i] = row[ks + 8 + i];
  return a;
}

// B 32x16 tile, storage is [n][k] (contiguous K per lane). element(k,n)=p[n*stride + kofs + k]
__device__ __forceinline__ v16h frag_b_nk(const _Float16* p, int stride, int kofs, int lane) {
  const int n  = lane & 15;
  const int kb = (lane >> 4) << 4;  // 0 or 16
  const _Float16* row = p + n * stride + kofs + kb;
  v16h b;
#pragma unroll
  for (int i = 0; i < 16; ++i) b[i] = row[i];
  return b;
}

// B 32x16 tile, storage is [k][n]. element(k,n)=p[k*stride + nofs + n]
__device__ __forceinline__ v16h frag_b_kn(const _Float16* p, int stride, int nofs, int lane) {
  const int n  = (lane & 15) + nofs;
  const int kb = (lane >> 4) << 4;
  v16h b;
#pragma unroll
  for (int i = 0; i < 16; ++i) b[i] = p[(kb + i) * stride + n];
  return b;
}

// ---------------- kernel 1: fused QKV projection ----------------
// Y = X @ W^T + b ; scatter f16 into [B][head][S][d]
// 128x128 block tile, 8 waves of 32x64 (8 WMMA / wave / K-step)
__global__ __launch_bounds__(256) void qkv_gemm_kernel(
    const float* __restrict__ X,
    const float* __restrict__ Wq, const float* __restrict__ Wk, const float* __restrict__ Wv,
    const float* __restrict__ bq, const float* __restrict__ bk, const float* __restrict__ bv,
    _Float16* __restrict__ qo, _Float16* __restrict__ ko, _Float16* __restrict__ vo) {
  __shared__ _Float16 As[128][40];   // [m][k]
  __shared__ _Float16 Bs[128][40];   // [n][k]
  const int tid  = threadIdx.x;
  const int lane = tid & 31;
  const int wave = tid >> 5;
  const int wr   = (wave >> 1) * 32;   // 0,32,64,96
  const int wc   = (wave & 1) * 64;    // 0,64
  const int row0 = blockIdx.x * 128;
  const int col0 = blockIdx.y * 128;
  const int z    = blockIdx.z;
  const float* W    = (z == 0) ? Wq : (z == 1) ? Wk : Wv;
  const float* bias = (z == 0) ? bq : (z == 1) ? bk : bv;
  _Float16*    dst  = (z == 0) ? qo : (z == 1) ? ko : vo;

  v8f acc[2][4] = {{ZERO8, ZERO8, ZERO8, ZERO8}, {ZERO8, ZERO8, ZERO8, ZERO8}};

  const int sr = tid >> 1;           // 0..127 (row for both A and B staging)
  const int sc = (tid & 1) * 16;     // 0,16

  for (int k0 = 0; k0 < HIDDEN; k0 += 32) {
    if (k0 + 32 < HIDDEN) {
      __builtin_prefetch((const void*)(X + (size_t)(row0 + sr) * HIDDEN + k0 + 32 + sc), 0, 1);
      __builtin_prefetch((const void*)(W + (size_t)(col0 + sr) * HIDDEN + k0 + 32 + sc), 0, 1);
    }
    {  // A tile 128x32, f32 -> f16
      const float4* src = (const float4*)(X + (size_t)(row0 + sr) * HIDDEN + k0 + sc);
      _Float16* d = &As[sr][sc];
#pragma unroll
      for (int q4 = 0; q4 < 4; ++q4) {
        float4 x = src[q4];
        d[q4 * 4 + 0] = (_Float16)x.x; d[q4 * 4 + 1] = (_Float16)x.y;
        d[q4 * 4 + 2] = (_Float16)x.z; d[q4 * 4 + 3] = (_Float16)x.w;
      }
    }
    {  // B tile: 128 output cols x 32 k, from W rows (W^T cols), f32 -> f16
      const float4* src = (const float4*)(W + (size_t)(col0 + sr) * HIDDEN + k0 + sc);
      _Float16* d = &Bs[sr][sc];
#pragma unroll
      for (int q4 = 0; q4 < 4; ++q4) {
        float4 x = src[q4];
        d[q4 * 4 + 0] = (_Float16)x.x; d[q4 * 4 + 1] = (_Float16)x.y;
        d[q4 * 4 + 2] = (_Float16)x.z; d[q4 * 4 + 3] = (_Float16)x.w;
      }
    }
    __syncthreads();
    v16h a0 = frag_a_lds(&As[wr][0], 40, 0, lane);
    v16h a1 = frag_a_lds(&As[wr + 16][0], 40, 0, lane);
#pragma unroll
    for (int j = 0; j < 4; ++j) {
      v16h b = frag_b_nk(&Bs[wc + j * 16][0], 40, 0, lane);
      acc[0][j] = WMMA(a0, b, acc[0][j]);
      acc[1][j] = WMMA(a1, b, acc[1][j]);
    }
    __syncthreads();
  }

  const int hi = lane >> 4;
  const int nn = lane & 15;
#pragma unroll
  for (int i = 0; i < 2; ++i) {
#pragma unroll
    for (int j = 0; j < 4; ++j) {
      const int col = col0 + wc + j * 16 + nn;
      const float bb = bias[col];
      const int head = col >> 6, dd = col & 63;
#pragma unroll
      for (int r = 0; r < 8; ++r) {
        const int row = row0 + wr + i * 16 + r + 8 * hi;
        const int bi = row >> 11, s = row & (SEQ - 1);
        const float val = acc[i][j][r] + bb;
        dst[(((size_t)bi * HEADS + head) * SEQ + s) * HEAD_DIM + dd] = (_Float16)val;
      }
    }
  }
}

// ---------------- kernel 2: RoPE in-place on q and k ----------------
__global__ __launch_bounds__(256) void rope_kernel(_Float16* __restrict__ q,
                                                   _Float16* __restrict__ k) {
  const int idx = blockIdx.x * blockDim.x + threadIdx.x;  // [0, 2*32*2048*32)
  const int j  = idx & 31;
  const int t  = idx >> 5;
  const int s  = t & (SEQ - 1);
  const int t2 = t >> 11;
  const int bh = t2 & 31;
  const int wh = t2 >> 5;
  _Float16* p = wh ? k : q;
  const size_t off = ((size_t)bh * SEQ + s) * HEAD_DIM;
  const float inv = __expf(-(float)j * 0.28782313663f);  // ln(10000)/32
  float sn, cs;
  __sincosf((float)s * inv, &sn, &cs);
  const float x1 = (float)p[off + j];
  const float x2 = (float)p[off + j + 32];
  p[off + j]      = (_Float16)(x1 * cs - x2 * sn);
  p[off + j + 32] = (_Float16)(x2 * cs + x1 * sn);
}

// ---------------- kernel 3: flash attention ----------------
// 4 waves x 16 query rows; 64-key chunks (16 WMMA per chunk)
__global__ __launch_bounds__(128) void attn_kernel(
    const _Float16* __restrict__ q, const _Float16* __restrict__ kk,
    const _Float16* __restrict__ vv, _Float16* __restrict__ ob) {
  __shared__ _Float16 Kt[64][72];      // [key][d]
  __shared__ _Float16 Vt[64][72];      // [key][d]
  __shared__ _Float16 Pb[4][16][72];   // per-wave P (16 x 64 keys) bounce
  const int tid  = threadIdx.x;
  const int lane = tid & 31;
  const int wave = tid >> 5;
  const int hi   = lane >> 4;
  const int nn   = lane & 15;
  const int bh   = blockIdx.y;
  const int b = bh >> 4, head = bh & 15;
  const size_t base = (size_t)bh * SEQ * HEAD_DIM;
  const int q0 = blockIdx.x * 64 + wave * 16;

  // q as two A fragments (K = d = 0..31, 32..63), pre-scaled by 1/sqrt(64)
  v16h aq0, aq1;
  {
    const _Float16* qp = q + base + (size_t)(q0 + nn) * HEAD_DIM;
    const int ks = hi * 16;
#pragma unroll
    for (int i = 0; i < 8; ++i) {
      aq0[i]     = (_Float16)((float)qp[ks + i] * 0.125f);
      aq0[8 + i] = (_Float16)((float)qp[8 + ks + i] * 0.125f);
      aq1[i]     = (_Float16)((float)qp[32 + ks + i] * 0.125f);
      aq1[8 + i] = (_Float16)((float)qp[40 + ks + i] * 0.125f);
    }
  }

  v8f o[4] = {ZERO8, ZERO8, ZERO8, ZERO8};
  float m_run[8], l_run[8];
#pragma unroll
  for (int r = 0; r < 8; ++r) { m_run[r] = -1e30f; l_run[r] = 0.f; }

  const int ky = tid >> 1;          // 0..63
  const int kx = (tid & 1) * 32;    // 0,32

  for (int kc = 0; kc < SEQ; kc += 64) {
    {  // stage K/V chunk (64 keys x 64 dims) via async global->LDS (f16 copy)
      const _Float16* ksrc = kk + base + (size_t)(kc + ky) * HEAD_DIM + kx;
      const _Float16* vsrc = vv + base + (size_t)(kc + ky) * HEAD_DIM + kx;
#pragma unroll
      for (int i = 0; i < 4; ++i) {
        async_cp16(ksrc + 8 * i, &Kt[ky][kx + 8 * i]);
        async_cp16(vsrc + 8 * i, &Vt[ky][kx + 8 * i]);
      }
      async_wait0();
    }
    __syncthreads();

    // logits: four 16x16 tiles over keys kc+16*t, K-dim = 64
    v8f st[4] = {ZERO8, ZERO8, ZERO8, ZERO8};
#pragma unroll
    for (int t = 0; t < 4; ++t) {
      st[t] = WMMA(aq0, frag_b_nk(&Kt[16 * t][0], 72, 0, lane), st[t]);
      st[t] = WMMA(aq1, frag_b_nk(&Kt[16 * t][0], 72, 32, lane), st[t]);
    }

    // online softmax; row M lives in 16 lanes of one half-wave
#pragma unroll
    for (int r = 0; r < 8; ++r) {
      float mx = fmaxf(fmaxf(st[0][r], st[1][r]), fmaxf(st[2][r], st[3][r]));
      mx = fmaxf(mx, __shfl_xor(mx, 1, 32));
      mx = fmaxf(mx, __shfl_xor(mx, 2, 32));
      mx = fmaxf(mx, __shfl_xor(mx, 4, 32));
      mx = fmaxf(mx, __shfl_xor(mx, 8, 32));
      const float mnew = fmaxf(m_run[r], mx);
      const float scl = __expf(m_run[r] - mnew);
      m_run[r] = mnew;
      float rs = 0.f;
#pragma unroll
      for (int t = 0; t < 4; ++t) {
        const float e = __expf(st[t][r] - mnew);
        rs += e;
        Pb[wave][r + 8 * hi][16 * t + nn] = (_Float16)e;
      }
      rs += __shfl_xor(rs, 1, 32);
      rs += __shfl_xor(rs, 2, 32);
      rs += __shfl_xor(rs, 4, 32);
      rs += __shfl_xor(rs, 8, 32);
      l_run[r] = l_run[r] * scl + rs;
      o[0][r] *= scl; o[1][r] *= scl; o[2][r] *= scl; o[3][r] *= scl;
    }
    __syncthreads();

    // O += P(16x64) * V(64x64)
    v16h ap0 = frag_a_lds(&Pb[wave][0][0], 72, 0, lane);
    v16h ap1 = frag_a_lds(&Pb[wave][0][0], 72, 32, lane);
#pragma unroll
    for (int j = 0; j < 4; ++j) {
      o[j] = WMMA(ap0, frag_b_kn(&Vt[0][0], 72, j * 16, lane), o[j]);
      o[j] = WMMA(ap1, frag_b_kn(&Vt[32][0], 72, j * 16, lane), o[j]);
    }
    __syncthreads();
  }

#pragma unroll
  for (int r = 0; r < 8; ++r) {
    const float rl = 1.0f / l_run[r];
    const int s = q0 + r + 8 * hi;
    const size_t orow = ((size_t)b * SEQ + s) * HIDDEN + head * HEAD_DIM;
#pragma unroll
    for (int j = 0; j < 4; ++j) ob[orow + j * 16 + nn] = (_Float16)(o[j][r] * rl);
  }
}

// ---------------- kernel 4: output projection ----------------
__global__ __launch_bounds__(256) void out_gemm_kernel(
    const _Float16* __restrict__ A, const float* __restrict__ W,
    const float* __restrict__ bias, float* __restrict__ out) {
  __shared__ _Float16 As[128][40];
  __shared__ _Float16 Bs[128][40];
  const int tid  = threadIdx.x;
  const int lane = tid & 31;
  const int wave = tid >> 5;
  const int wr   = (wave >> 1) * 32;
  const int wc   = (wave & 1) * 64;
  const int row0 = blockIdx.x * 128;
  const int col0 = blockIdx.y * 128;

  v8f acc[2][4] = {{ZERO8, ZERO8, ZERO8, ZERO8}, {ZERO8, ZERO8, ZERO8, ZERO8}};

  const int sr = tid >> 1;
  const int sc = (tid & 1) * 16;

  for (int k0 = 0; k0 < HIDDEN; k0 += 32) {
    if (k0 + 32 < HIDDEN) {
      __builtin_prefetch((const void*)(W + (size_t)(col0 + sr) * HIDDEN + k0 + 32 + sc), 0, 1);
    }
    {  // A tile already f16: async copy straight into LDS
      const _Float16* src = A + (size_t)(row0 + sr) * HIDDEN + k0 + sc;
      async_cp16(src, &As[sr][sc]);
      async_cp16(src + 8, &As[sr][sc + 8]);
    }
    {  // B tile f32 -> f16
      const float4* src = (const float4*)(W + (size_t)(col0 + sr) * HIDDEN + k0 + sc);
      _Float16* d = &Bs[sr][sc];
#pragma unroll
      for (int q4 = 0; q4 < 4; ++q4) {
        float4 x = src[q4];
        d[q4 * 4 + 0] = (_Float16)x.x; d[q4 * 4 + 1] = (_Float16)x.y;
        d[q4 * 4 + 2] = (_Float16)x.z; d[q4 * 4 + 3] = (_Float16)x.w;
      }
    }
    async_wait0();
    __syncthreads();
    v16h a0 = frag_a_lds(&As[wr][0], 40, 0, lane);
    v16h a1 = frag_a_lds(&As[wr + 16][0], 40, 0, lane);
#pragma unroll
    for (int j = 0; j < 4; ++j) {
      v16h b = frag_b_nk(&Bs[wc + j * 16][0], 40, 0, lane);
      acc[0][j] = WMMA(a0, b, acc[0][j]);
      acc[1][j] = WMMA(a1, b, acc[1][j]);
    }
    __syncthreads();
  }

  const int hi = lane >> 4;
  const int nn = lane & 15;
#pragma unroll
  for (int i = 0; i < 2; ++i) {
#pragma unroll
    for (int j = 0; j < 4; ++j) {
      const int col = col0 + wc + j * 16 + nn;
      const float bb = bias[col];
#pragma unroll
      for (int r = 0; r < 8; ++r) {
        const int row = row0 + wr + i * 16 + r + 8 * hi;
        out[(size_t)row * HIDDEN + col] = acc[i][j][r] + bb;
      }
    }
  }
}

extern "C" void kernel_launch(void* const* d_in, const int* in_sizes, int n_in,
                              void* d_out, int out_size, void* d_ws, size_t ws_size,
                              hipStream_t stream) {
  const float* X  = (const float*)d_in[0];
  const float* Wq = (const float*)d_in[1];
  const float* bq = (const float*)d_in[2];
  const float* Wk = (const float*)d_in[3];
  const float* bk = (const float*)d_in[4];
  const float* Wv = (const float*)d_in[5];
  const float* bv = (const float*)d_in[6];
  const float* Wo = (const float*)d_in[7];
  const float* bo = (const float*)d_in[8];
  float* out = (float*)d_out;

  _Float16* ws  = (_Float16*)d_ws;
  _Float16* qws = ws;
  _Float16* kws = ws + PER_T;
  _Float16* vws = ws + 2 * PER_T;
  _Float16* obf = ws + 3 * PER_T;   // [B][S][H] f16, total 32 MB of ws

  qkv_gemm_kernel<<<dim3(NROWS / 128, HIDDEN / 128, 3), 256, 0, stream>>>(
      X, Wq, Wk, Wv, bq, bk, bv, qws, kws, vws);
  rope_kernel<<<(2 * BATCH * HEADS * SEQ * 32) / 256, 256, 0, stream>>>(qws, kws);
  attn_kernel<<<dim3(SEQ / 64, BATCH * HEADS), 128, 0, stream>>>(qws, kws, vws, obf);
  out_gemm_kernel<<<dim3(NROWS / 128, HIDDEN / 128), 256, 0, stream>>>(obf, Wo, bo, out);
}
